// LoRALinear_75548474736691
// MI455X (gfx1250) — matrix operand order repaired and checked by reading, not compiled, not run on hardware
//
#include <hip/hip_runtime.h>

// out = X@W^T + bias + 2.0*((X@A^T)@B^T)
// M=8192, N=4096, K=4096, R=16. Compute-bound -> bf16 WMMA path.
// 256 thr (8 waves), 128x128 tile, K-tile 32, double-buffered LDS pipeline,
// LoRA x@A^T fused as extra WMMA n-tile; v_perm_b32 bf16 packing.

typedef __attribute__((ext_vector_type(16))) __bf16 v16bf;
typedef __attribute__((ext_vector_type(8)))  float  v8f;

#define K_DIM 4096
#define N_DIM 4096
#define SCALE 2.0f   // ALPHA/R = 32/16

// Pack two f32 -> two bf16 (round-half-up) with a single v_perm_b32.
__device__ __forceinline__ unsigned pk2(float lo, float hi) {
  return __builtin_amdgcn_perm(__float_as_uint(hi) + 0x8000u,
                               __float_as_uint(lo) + 0x8000u, 0x07060302u);
}
__device__ __forceinline__ uint4 pk8(float4 a, float4 b) {
  return make_uint4(pk2(a.x, a.y), pk2(a.z, a.w), pk2(b.x, b.y), pk2(b.z, b.w));
}

union FragAB { v16bf v; uint4 q[2]; };

__global__ __launch_bounds__(256, 2)
void lora_gemm_kernel(const float* __restrict__ X, const float* __restrict__ W,
                      const float* __restrict__ bias, const float* __restrict__ Al,
                      const float* __restrict__ Bl, float* __restrict__ out)
{
  __shared__ unsigned short sX[2][128 * 40];  // 128 rows x 32 bf16, pad 40 (80B, 16B-aligned)
  __shared__ unsigned short sW[2][128 * 40];
  __shared__ float sXA[128 * 16];

  const int tid  = threadIdx.x;
  const int lane = tid & 31;
  const int wave = tid >> 5;
  const int wm   = wave & 3;     // 4 waves over M (32 rows each)
  const int wn   = wave >> 2;    // 2 waves over N (64 cols each)
  const int l16  = lane & 15;
  const bool hi  = lane >= 16;

  const int m0 = blockIdx.y * 128;
  const int n0 = blockIdx.x * 128;

  // Staging role: thread t -> row t/2, half t&1 (16 consecutive k)
  const int srow  = tid >> 1;
  const int shalf = tid & 1;
  const float* gXb = X + (size_t)(m0 + srow) * K_DIM + shalf * 16;
  const float* gWb = W + (size_t)(n0 + srow) * K_DIM + shalf * 16;
  const int soff = srow * 40 + shalf * 16;

  v8f acc[2][4];
  v8f xacc[2];
  #pragma unroll
  for (int mt = 0; mt < 2; ++mt) {
    #pragma unroll
    for (int i = 0; i < 8; ++i) xacc[mt][i] = 0.0f;
    #pragma unroll
    for (int nt = 0; nt < 4; ++nt)
      #pragma unroll
      for (int i = 0; i < 8; ++i) acc[mt][nt][i] = 0.0f;
  }

  // ---- prologue: stage k=0 into buffer 0 ----
  {
    const float4* gx = (const float4*)gXb;
    const float4* gw = (const float4*)gWb;
    float4 x0 = gx[0], x1 = gx[1], x2 = gx[2], x3 = gx[3];
    float4 w0 = gw[0], w1 = gw[1], w2 = gw[2], w3 = gw[3];
    *(uint4*)&sX[0][soff]     = pk8(x0, x1);
    *(uint4*)&sX[0][soff + 8] = pk8(x2, x3);
    *(uint4*)&sW[0][soff]     = pk8(w0, w1);
    *(uint4*)&sW[0][soff + 8] = pk8(w2, w3);
  }
  __syncthreads();

  const int c0 = hi ? 16 : 0;   // A-frag element offset
  const int kb = hi ? 16 : 0;   // B-frag element offset

  for (int k0 = 0; k0 < K_DIM; k0 += 32) {
    const int cur = (k0 >> 5) & 1;
    const unsigned short* sXc = sX[cur];
    const unsigned short* sWc = sW[cur];
    unsigned short* dX = sX[cur ^ 1];
    unsigned short* dW = sW[cur ^ 1];

    // ---- prefetch next K-slab into registers (overlaps WMMA below) ----
    const bool more = (k0 + 32) < K_DIM;
    float4 nx0, nx1, nx2, nx3, nw0, nw1, nw2, nw3;
    if (more) {
      const float4* gx = (const float4*)(gXb + k0 + 32);
      const float4* gw = (const float4*)(gWb + k0 + 32);
      nx0 = gx[0]; nx1 = gx[1]; nx2 = gx[2]; nx3 = gx[3];
      nw0 = gw[0]; nw1 = gw[1]; nw2 = gw[2]; nw3 = gw[3];
    }

    // ---- A fragments (16x32 bf16): two 16B chunks per lane ----
    FragAB a[2];
    #pragma unroll
    for (int mt = 0; mt < 2; ++mt) {
      const unsigned short* p = &sXc[(wm * 32 + mt * 16 + l16) * 40 + c0];
      a[mt].q[0] = *(const uint4*)p;
      a[mt].q[1] = *(const uint4*)(p + 8);
    }
    // ---- B fragments (32x16 bf16): one contiguous 32B chunk per lane ----
    FragAB b[4];
    #pragma unroll
    for (int nt = 0; nt < 4; ++nt) {
      const unsigned short* p = &sWc[(wn * 64 + nt * 16 + l16) * 40 + kb];
      b[nt].q[0] = *(const uint4*)p;
      b[nt].q[1] = *(const uint4*)(p + 8);
    }

    #pragma unroll
    for (int mt = 0; mt < 2; ++mt)
      #pragma unroll
      for (int nt = 0; nt < 4; ++nt)
        acc[mt][nt] = __builtin_amdgcn_wmma_f32_16x16x32_bf16(
            false, a[mt].v, false, b[nt].v, (short)0, acc[mt][nt], false, false);

    // ---- LoRA: xa += X_tile @ Al_chunk^T (wave-uniform branch, EXEC stays full) ----
    if (wn == 0) {
      FragAB lb;  // B-frag from LoRA A [16 x K] row-major: lane l16 = r, contiguous k
      const float4* gA = (const float4*)(Al + (size_t)l16 * K_DIM + k0 + (hi ? 16 : 0));
      float4 v0 = gA[0], v1 = gA[1], v2 = gA[2], v3 = gA[3];
      lb.q[0] = pk8(v0, v1);
      lb.q[1] = pk8(v2, v3);
      #pragma unroll
      for (int mt = 0; mt < 2; ++mt)
        xacc[mt] = __builtin_amdgcn_wmma_f32_16x16x32_bf16(
            false, a[mt].v, false, lb.v, (short)0, xacc[mt], false, false);
    }

    // ---- convert + store prefetched slab into the other buffer ----
    if (more) {
      *(uint4*)&dX[soff]     = pk8(nx0, nx1);
      *(uint4*)&dX[soff + 8] = pk8(nx2, nx3);
      *(uint4*)&dW[soff]     = pk8(nw0, nw1);
      *(uint4*)&dW[soff + 8] = pk8(nw2, nw3);
    }
    __syncthreads();
  }

  // ---- publish xa to LDS (C layout: vgpr i -> m = i + (hi?8:0), col = r = l16) ----
  if (wn == 0) {
    #pragma unroll
    for (int mt = 0; mt < 2; ++mt)
      #pragma unroll
      for (int i = 0; i < 8; ++i) {
        int m = wm * 32 + mt * 16 + i + (hi ? 8 : 0);
        sXA[m * 16 + l16] = xacc[mt][i];
      }
  }
  __syncthreads();

  // ---- epilogue: + bias + SCALE * xa . Bl[n,:] ----
  float  bn[4];
  float4 br[4][4];
  #pragma unroll
  for (int nt = 0; nt < 4; ++nt) {
    int n = n0 + wn * 64 + nt * 16 + l16;
    bn[nt] = bias[n];
    const float4* pb = (const float4*)(Bl + (size_t)n * 16);
    br[nt][0] = pb[0]; br[nt][1] = pb[1]; br[nt][2] = pb[2]; br[nt][3] = pb[3];
  }
  #pragma unroll
  for (int mt = 0; mt < 2; ++mt) {
    #pragma unroll
    for (int i = 0; i < 8; ++i) {
      int ml = wm * 32 + mt * 16 + i + (hi ? 8 : 0);
      const float4* pxa = (const float4*)&sXA[ml * 16];  // broadcast across lanes 0-15
      float4 x0 = pxa[0], x1 = pxa[1], x2 = pxa[2], x3 = pxa[3];
      #pragma unroll
      for (int nt = 0; nt < 4; ++nt) {
        float d = x0.x*br[nt][0].x + x0.y*br[nt][0].y + x0.z*br[nt][0].z + x0.w*br[nt][0].w
                + x1.x*br[nt][1].x + x1.y*br[nt][1].y + x1.z*br[nt][1].z + x1.w*br[nt][1].w
                + x2.x*br[nt][2].x + x2.y*br[nt][2].y + x2.z*br[nt][2].z + x2.w*br[nt][2].w
                + x3.x*br[nt][3].x + x3.y*br[nt][3].y + x3.z*br[nt][3].z + x3.w*br[nt][3].w;
        int n = n0 + wn * 64 + nt * 16 + l16;
        out[(size_t)(m0 + ml) * N_DIM + n] = acc[mt][nt][i] + bn[nt] + SCALE * d;
      }
    }
  }
}

extern "C" void kernel_launch(void* const* d_in, const int* in_sizes, int n_in,
                              void* d_out, int out_size, void* d_ws, size_t ws_size,
                              hipStream_t stream) {
  const float* x  = (const float*)d_in[0];
  const float* w  = (const float*)d_in[1];
  const float* bi = (const float*)d_in[2];
  const float* Al = (const float*)d_in[3];
  const float* Bl = (const float*)d_in[4];
  float* out = (float*)d_out;
  dim3 grid(N_DIM / 128, 8192 / 128);   // 32 x 64 blocks
  lora_gemm_kernel<<<grid, dim3(256), 0, stream>>>(x, w, bi, Al, Bl, out);
}
